// SequenceRouter_71442486001885
// MI455X (gfx1250) — compile-verified, matrix operand order checked by reading, not run on hardware
//
#include <hip/hip_runtime.h>
#include <hip/hip_fp16.h>

// ============================================================================
// SequenceRouter for MI455X (gfx1250, wave32, WMMA).
// Pipeline:
//   conv1 (VALU) -> im2col -> conv2 GEMM (WMMA f16->f32) -> bias+relu(f16)
//   -> proj GEMM (WMMA) -> emb-capsule maxout/squash/LN (VALU)
//   -> u_hat0 GEMM (WMMA, K=8 padded to 32) -> sequential routing (4 WGs,
//      layer0 slice staged in LDS via async global->LDS; layer1 prefetched)
//   -> LN128 -> u_hat1 GEMM (WMMA) -> routing -> LN512 -> length+LN32 -> out
// Requires ws_size >= ~432 MB (u_hat [2048,80,512] f32 dominates).
// ============================================================================

typedef __attribute__((ext_vector_type(16))) _Float16 v16h;
typedef __attribute__((ext_vector_type(8)))  float    v8f;
typedef __attribute__((ext_vector_type(4)))  int      v4i;
typedef v4i __attribute__((address_space(1)))* gptr_v4i;  // global int4*
typedef v4i __attribute__((address_space(3)))* lptr_v4i;  // LDS int4*

#define SQ_EPS 1e-9f
#define LN_EPS 1e-3f
#define CDIV(a, b) (((a) + (b) - 1) / (b))

#if defined(__has_builtin)
#if __has_builtin(__builtin_amdgcn_global_load_async_to_lds_b128)
#define HAVE_ASYNC_LDS 1
#endif
#if __has_builtin(__builtin_amdgcn_s_wait_asynccnt)
#define HAVE_WAIT_ASYNC 1
#endif
#endif

// ---- f16 WMMA fragment loader: row-major [.., K] source ---------------------
// 16-bit A/B 16x32 layout (ISA 7.12.2): lane holds row (lane&15);
// VGPR j holds f16 pair at K = (j>=4?16:0) + (j&3)*2 + (lane>=16)*8.
static __device__ __forceinline__ v16h load_frag16(const _Float16* rowp, int k0, int lane) {
  union { unsigned u[8]; v16h h; } r;
  const _Float16* p = rowp + k0 + ((lane >> 4) & 1) * 8;
#pragma unroll
  for (int j = 0; j < 8; ++j) {
    const int kb = ((j & 4) << 2) + ((j & 3) << 1);  // 0,2,4,6,16,18,20,22
    r.u[j] = *(const unsigned*)(p + kb);
  }
  return r.h;
}

// ---- generic WMMA GEMM: C[M,N] = A[M,K] * BT[N,K]^T  (K % 32 == 0) ----------
__global__ void gemm_wmma_f16(const _Float16* __restrict__ A,
                              const _Float16* __restrict__ BT,
                              float* __restrict__ C,
                              int M, int N, int K, int ntiles, int total_tiles) {
  const int lane = threadIdx.x & 31;
  const int wave = threadIdx.x >> 5;
  const int tile = blockIdx.x * (blockDim.x >> 5) + wave;
  if (tile >= total_tiles) return;
  const int nt = tile % ntiles;
  const int mt = tile / ntiles;
  const _Float16* arow = A + (size_t)(mt * 16 + (lane & 15)) * K;
  const _Float16* brow = BT + (size_t)(nt * 16 + (lane & 15)) * K;
  v8f c = {};
  for (int k0 = 0; k0 < K; k0 += 32) {
    v16h a = load_frag16(arow, k0, lane);
    v16h b = load_frag16(brow, k0, lane);
    c = __builtin_amdgcn_wmma_f32_16x16x32_f16(false, a, false, b, (short)0, c,
                                               false, false);
  }
  const int n = nt * 16 + (lane & 15);
  const int mbase = mt * 16 + ((lane >> 4) << 3);
#pragma unroll
  for (int r = 0; r < 8; ++r)
    C[(size_t)(mbase + r) * N + n] = c[r];
}

// ---- u_hat GEMM: per input-capsule n, U[bt,n,op] = emb_win[bt,8] . W[n,8,op] + B
// K=8 zero-padded to 32 inside the fragment (only lanes 0..15, VGPRs 0..3 live).
// emb window: x[b,t, n=k*16+i, q] = emb[b, t+k-2, i, q] (zero-padded in t).
__global__ void uhat_wmma(const _Float16* __restrict__ embh,  // [B*T,16,8]
                          const _Float16* __restrict__ Wh,    // [80,OP,8]
                          const float* __restrict__ Bias,     // [80,OP]
                          float* __restrict__ U,              // [B*T,80,OP]
                          int OP) {
  const int lane = threadIdx.x & 31;
  const int nt = blockIdx.x;  // OP/16
  const int mt = blockIdx.y;  // 128 tiles over B*T=2048
  const int n  = blockIdx.z;  // 80
  const int kwin = n >> 4, icap = n & 15;
  const int half = lane >> 4;
  const int m15 = lane & 15;
  union { unsigned u[8]; v16h h; } af, bf;
#pragma unroll
  for (int j = 0; j < 8; ++j) { af.u[j] = 0u; bf.u[j] = 0u; }
  if (half == 0) {
    const int bt = mt * 16 + m15;
    const int b = bt >> 9, t = bt & 511;
    const int ts = t + kwin - 2;
    if ((unsigned)ts < 512u) {
      const _Float16* ap = embh + (size_t)(((b << 9) + ts) * 16 + icap) * 8;
#pragma unroll
      for (int j = 0; j < 4; ++j) af.u[j] = *(const unsigned*)(ap + 2 * j);
    }
    const _Float16* bp = Wh + (size_t)(n * OP + nt * 16 + m15) * 8;
#pragma unroll
    for (int j = 0; j < 4; ++j) bf.u[j] = *(const unsigned*)(bp + 2 * j);
  }
  v8f c = {};
  c = __builtin_amdgcn_wmma_f32_16x16x32_f16(false, af.h, false, bf.h, (short)0,
                                             c, false, false);
  const int col = nt * 16 + m15;
  const float bias = Bias[n * OP + col];
  const int mbase = mt * 16 + (half << 3);
#pragma unroll
  for (int r = 0; r < 8; ++r)
    U[((size_t)(mbase + r) * 80 + n) * OP + col] = c[r] + bias;
}

// ---- conv1 3x3x1->64 stride2 + relu, f16 out (SAME: pad_lo=0, pad_hi=1) -----
__global__ void conv1_relu(const float* __restrict__ in,      // [4,2048,80,1]
                           const float* __restrict__ w,       // [3,3,1,64]
                           const float* __restrict__ bias,
                           _Float16* __restrict__ out) {      // [4,1024,40,64]
  long idx = (long)blockIdx.x * blockDim.x + threadIdx.x;
  if (idx >= 4L * 1024 * 40 * 64) return;
  int c = idx & 63;
  int q = (int)(idx >> 6);
  int f = q % 40;
  int t = (q / 40) % 1024;
  int b = q / (40 * 1024);
  float acc = bias[c];
#pragma unroll
  for (int kh = 0; kh < 3; ++kh) {
    int r = 2 * t + kh;
    if (r >= 2048) continue;
#pragma unroll
    for (int kw = 0; kw < 3; ++kw) {
      int cc = 2 * f + kw;
      if (cc >= 80) continue;
      acc += in[(size_t)(b * 2048 + r) * 80 + cc] * w[(kh * 3 + kw) * 64 + c];
    }
  }
  out[idx] = (_Float16)fmaxf(acc, 0.f);
}

// ---- im2col for conv2 (3x3x64, stride2, SAME pad_lo=0) ----------------------
__global__ void im2col2(const _Float16* __restrict__ x1,      // [4,1024,40,64]
                        _Float16* __restrict__ acol) {        // [40960,576]
  long idx = (long)blockIdx.x * blockDim.x + threadIdx.x;
  if (idx >= 40960L * 576) return;
  int kk = (int)(idx % 576);
  long m = idx / 576;
  int c = kk & 63;
  int kw = (kk >> 6) % 3;
  int kh = (kk >> 6) / 3;
  int f = (int)(m % 20);
  int t = (int)((m / 20) % 512);
  int b = (int)(m / (20 * 512));
  int r = 2 * t + kh, cc = 2 * f + kw;
  _Float16 v = (_Float16)0.f;
  if (r < 1024 && cc < 40)
    v = x1[((size_t)(b * 1024 + r) * 40 + cc) * 64 + c];
  acol[idx] = v;
}

// ---- elementwise epilogues / casts ------------------------------------------
__global__ void bias_relu_h(const float* __restrict__ x, const float* __restrict__ b,
                            _Float16* __restrict__ y, long n, int C) {
  long i = (long)blockIdx.x * blockDim.x + threadIdx.x;
  if (i >= n) return;
  y[i] = (_Float16)fmaxf(x[i] + b[i % C], 0.f);
}
__global__ void bias_add_f(const float* __restrict__ x, const float* __restrict__ b,
                           float* __restrict__ y, long n, int C) {
  long i = (long)blockIdx.x * blockDim.x + threadIdx.x;
  if (i >= n) return;
  y[i] = x[i] + b[i % C];
}
__global__ void cast_h(const float* __restrict__ x, _Float16* __restrict__ y, long n) {
  long i = (long)blockIdx.x * blockDim.x + threadIdx.x;
  if (i < n) y[i] = (_Float16)x[i];
}
__global__ void transpose_c2(const float* __restrict__ w, _Float16* __restrict__ bt) {
  int i = blockIdx.x * blockDim.x + threadIdx.x;   // 64*576
  if (i >= 64 * 576) return;
  int co = i / 576, k = i % 576;                   // w[(kh,kw,ci),co]
  bt[co * 576 + k] = (_Float16)w[(size_t)k * 64 + co];
}
__global__ void transpose_proj(const float* __restrict__ w, _Float16* __restrict__ bt) {
  int i = blockIdx.x * blockDim.x + threadIdx.x;   // 16*1280
  if (i >= 16 * 1280) return;
  int n = i / 1280, k = i % 1280;
  bt[n * 1280 + k] = (_Float16)w[(size_t)k * 16 + n];
}

// ---- emb capsule: maxout(3x3 conv x2) * tmask -> squash(8) -> LN(128) -> f16
__global__ void embcaps(const float* __restrict__ embp,   // [4,512,16]
                        const float* __restrict__ w0, const float* __restrict__ b0,
                        const float* __restrict__ w1, const float* __restrict__ b1,
                        const float* __restrict__ lng, const float* __restrict__ lnb,
                        const int* __restrict__ lens,
                        _Float16* __restrict__ out) {      // [4,512,16,8]
  int r = blockIdx.x * blockDim.x + threadIdx.x;
  if (r >= 2048) return;
  int b = r >> 9, t = r & 511;
  float win[3][18];
#pragma unroll
  for (int dh = 0; dh < 3; ++dh) {
    int tt = t + dh - 1;
#pragma unroll
    for (int c = 0; c < 18; ++c) {
      int ii = c - 1;
      win[dh][c] = ((unsigned)tt < 512u && (unsigned)ii < 16u)
                       ? embp[(size_t)((b << 9) + tt) * 16 + ii] : 0.f;
    }
  }
  float tm = (t < ((lens[b] + 3) >> 2)) ? 1.f : 0.f;
  float tmp[128];
#pragma unroll
  for (int i = 0; i < 16; ++i) {
#pragma unroll
    for (int j = 0; j < 8; ++j) {
      float a0 = b0[j], a1 = b1[j];
#pragma unroll
      for (int dh = 0; dh < 3; ++dh)
#pragma unroll
        for (int dw = 0; dw < 3; ++dw) {
          float x = win[dh][i + dw];
          a0 += x * w0[(dh * 3 + dw) * 8 + j];
          a1 += x * w1[(dh * 3 + dw) * 8 + j];
        }
      tmp[i * 8 + j] = fmaxf(a0, a1) * tm;
    }
    float sq = 0.f;
#pragma unroll
    for (int j = 0; j < 8; ++j) { float x = tmp[i * 8 + j]; sq += x * x; }
    float sc = (sq / (1.f + sq)) * rsqrtf(sq + SQ_EPS);
#pragma unroll
    for (int j = 0; j < 8; ++j) tmp[i * 8 + j] *= sc;
  }
  float m = 0.f;
#pragma unroll
  for (int i = 0; i < 128; ++i) m += tmp[i];
  m *= (1.f / 128.f);
  float v = 0.f;
#pragma unroll
  for (int i = 0; i < 128; ++i) { float d = tmp[i] - m; v += d * d; }
  float inv = rsqrtf(v * (1.f / 128.f) + LN_EPS);
#pragma unroll
  for (int i = 0; i < 128; ++i)
    out[(size_t)r * 128 + i] = (_Float16)((tmp[i] - m) * inv * lng[i] + lnb[i]);
}

// ---- sequential dynamic routing: one WG per batch, 512-step scan ------------
// stage!=0 (layer0, 80*OP*4 == 40KB): slice is staged into LDS each step with
// GLOBAL_LOAD_ASYNC_TO_LDS_B128 (ASYNCcnt) when available, else float4 copy.
// stage==0 (layer1, 164KB slice): direct global reads + prefetch of t+1 slice.
__global__ void route_scan(const float* __restrict__ U,  // [B*T,80,CO*CD]
                           float* __restrict__ VS,       // [B*T,CO,CD]
                           int CO, int CD, int maskFirst, int stage) {
  const int b = blockIdx.x;
  const int tid = threadIdx.x;
  const int NT = blockDim.x;
  const int OP = CO * CD;
  const int CD4 = CD >> 2;
  __shared__ float v[512];
  __shared__ float lg[2560];
  __shared__ float sb[512];
  __shared__ float ustage[10240];  // 40KB: layer-0 u_t slice
  for (int i = tid; i < OP; i += NT) v[i] = 0.f;
  __syncthreads();
  for (int t = 0; t < 512; ++t) {
    const float* u = U + (size_t)((b << 9) + t) * 80 * OP;
    // prefetch next timestep's u_hat slice (global_prefetch_b8) into cache
    if (t + 1 < 512) {
      const float* un = u + (size_t)80 * OP;
      const int lines = (80 * OP) >> 5;  // 128B cachelines
      for (int l = tid; l < lines; l += NT)
        __builtin_prefetch(un + (l << 5), 0, 1);
    }
    const float* us = u;
    if (stage) {
      const int chunks = (80 * OP) >> 2;  // float4 chunks (2560)
      for (int i4 = tid; i4 < chunks; i4 += NT) {
#if defined(HAVE_ASYNC_LDS)
        __builtin_amdgcn_global_load_async_to_lds_b128(
            (gptr_v4i)(u + (i4 << 2)), (lptr_v4i)(ustage + (i4 << 2)), 0, 0);
#else
        ((float4*)ustage)[i4] = ((const float4*)u)[i4];
#endif
      }
#if defined(HAVE_ASYNC_LDS)
#if defined(HAVE_WAIT_ASYNC)
      __builtin_amdgcn_s_wait_asynccnt(0);
#else
      asm volatile("s_wait_asynccnt 0" ::: "memory");
#endif
#endif
      __syncthreads();
      us = ustage;
    }
    // logits[n,o] = <u_t[n,o,:], v[o,:]> (+ mask on o==0), float4 dot
    for (int idx = tid; idx < 80 * CO; idx += NT) {
      int n = idx / CO, o = idx - n * CO;
      const float4* up = (const float4*)(us + (size_t)(n * CO + o) * CD);
      const float4* vp = (const float4*)(v + o * CD);
      float acc = 0.f;
      for (int p = 0; p < CD4; ++p) {
        float4 a = up[p], w = vp[p];
        acc += a.x * w.x + a.y * w.y + a.z * w.z + a.w * w.w;
      }
      if (maskFirst && o == 0) acc -= 1e9f;
      lg[idx] = acc;
    }
    __syncthreads();
    // softmax over o, per n
    for (int n = tid; n < 80; n += NT) {
      float mx = -3.4e38f;
      for (int o = 0; o < CO; ++o) mx = fmaxf(mx, lg[n * CO + o]);
      float s = 0.f;
      for (int o = 0; o < CO; ++o) {
        float e = __expf(lg[n * CO + o] - mx);
        lg[n * CO + o] = e;
        s += e;
      }
      float inv = 1.f / s;
      for (int o = 0; o < CO; ++o) lg[n * CO + o] *= inv;
    }
    __syncthreads();
    // s[o,p0..p0+3] = sum_n c[n,o] * u_t[n,o,p0..p0+3]  (float4 accumulate)
    for (int idx = tid; idx < (OP >> 2); idx += NT) {
      int o = idx / CD4, p4 = idx - o * CD4;
      float4 acc = make_float4(0.f, 0.f, 0.f, 0.f);
      for (int n = 0; n < 80; ++n) {
        float c = lg[n * CO + o];
        float4 a = ((const float4*)(us + (size_t)(n * CO + o) * CD))[p4];
        acc.x += c * a.x; acc.y += c * a.y; acc.z += c * a.z; acc.w += c * a.w;
      }
      ((float4*)sb)[idx] = acc;
    }
    __syncthreads();
    // squash -> v, emit v_t
    for (int o = tid; o < CO; o += NT) {
      float sq = 0.f;
      for (int p = 0; p < CD; ++p) { float x = sb[o * CD + p]; sq += x * x; }
      float sc = (sq / (1.f + sq)) * rsqrtf(sq + SQ_EPS);
      float* vo = v + o * CD;
      float* op = VS + ((size_t)((b << 9) + t) * CO + o) * CD;
      for (int p = 0; p < CD; ++p) {
        float val = sb[o * CD + p] * sc;
        vo[p] = val;
        op[p] = val;
      }
    }
    __syncthreads();
  }
}

// ---- row LayerNorm (optional f32/f16 outputs) -------------------------------
__global__ void ln_rows(const float* __restrict__ X, const float* __restrict__ g,
                        const float* __restrict__ bb, float* __restrict__ Yf,
                        _Float16* __restrict__ Yh, int R, int L) {
  int r = blockIdx.x * blockDim.x + threadIdx.x;
  if (r >= R) return;
  const float* x = X + (size_t)r * L;
  float m = 0.f;
  for (int i = 0; i < L; ++i) m += x[i];
  m /= (float)L;
  float v = 0.f;
  for (int i = 0; i < L; ++i) { float d = x[i] - m; v += d * d; }
  float inv = rsqrtf(v / (float)L + LN_EPS);
  for (int i = 0; i < L; ++i) {
    float y = (x[i] - m) * inv * g[i] + bb[i];
    if (Yf) Yf[(size_t)r * L + i] = y;
    if (Yh) Yh[(size_t)r * L + i] = (_Float16)y;
  }
}

// ---- capsule lengths + final LN(32) -> d_out --------------------------------
__global__ void final_len_ln(const float* __restrict__ X,  // [2048,32,16] (post-LN)
                             const float* __restrict__ g, const float* __restrict__ bb,
                             float* __restrict__ out) {     // [2048,32]
  int r = blockIdx.x * blockDim.x + threadIdx.x;
  if (r >= 2048) return;
  const float* x = X + (size_t)r * 512;
  float len[32];
#pragma unroll
  for (int o = 0; o < 32; ++o) {
    float sq = 0.f;
#pragma unroll
    for (int p = 0; p < 16; ++p) { float v = x[o * 16 + p]; sq += v * v; }
    len[o] = sqrtf(sq + SQ_EPS);
  }
  float m = 0.f;
#pragma unroll
  for (int o = 0; o < 32; ++o) m += len[o];
  m *= (1.f / 32.f);
  float v = 0.f;
#pragma unroll
  for (int o = 0; o < 32; ++o) { float d = len[o] - m; v += d * d; }
  float inv = rsqrtf(v * (1.f / 32.f) + LN_EPS);
#pragma unroll
  for (int o = 0; o < 32; ++o)
    out[(size_t)r * 32 + o] = (len[o] - m) * inv * g[o] + bb[o];
}

// ============================================================================
extern "C" void kernel_launch(void* const* d_in, const int* in_sizes, int n_in,
                              void* d_out, int out_size, void* d_ws, size_t ws_size,
                              hipStream_t stream) {
  (void)in_sizes; (void)n_in; (void)out_size; (void)ws_size;
  const float* inputs  = (const float*)d_in[0];
  const int*   lens    = (const int*)d_in[1];
  const float* conv1_w = (const float*)d_in[2];
  const float* conv1_b = (const float*)d_in[3];
  const float* conv2_w = (const float*)d_in[4];
  const float* conv2_b = (const float*)d_in[5];
  const float* proj_w  = (const float*)d_in[6];
  const float* proj_b  = (const float*)d_in[7];
  const float* ec0_w   = (const float*)d_in[8];
  const float* ec0_b   = (const float*)d_in[9];
  const float* ec1_w   = (const float*)d_in[10];
  const float* ec1_b   = (const float*)d_in[11];
  const float* lni_g   = (const float*)d_in[12];
  const float* lni_b   = (const float*)d_in[13];
  const float* W0      = (const float*)d_in[14];
  const float* B0      = (const float*)d_in[15];
  const float* W1      = (const float*)d_in[16];
  const float* B1      = (const float*)d_in[17];
  const float* lnm0_g  = (const float*)d_in[18];
  const float* lnm0_b  = (const float*)d_in[19];
  const float* lnm1_g  = (const float*)d_in[20];
  const float* lnm1_b  = (const float*)d_in[21];
  const float* lno_g   = (const float*)d_in[22];
  const float* lno_b   = (const float*)d_in[23];
  float* out = (float*)d_out;

  char* ws = (char*)d_ws;
  size_t off = 0;
  auto carve = [&](size_t bytes) -> void* {
    void* p = ws + off;
    off = (off + bytes + 255) & ~(size_t)255;
    return p;
  };
  _Float16* x1h   = (_Float16*)carve(4L * 1024 * 40 * 64 * 2);   // conv1 out f16
  _Float16* acol  = (_Float16*)carve(40960L * 576 * 2);          // im2col
  float*    c2    = (float*)carve(40960L * 64 * 4);              // conv2 raw
  _Float16* x2h   = (_Float16*)carve(2048L * 1280 * 2);          // relu f16
  float*    projC = (float*)carve(2048L * 16 * 4);
  float*    embp  = (float*)carve(2048L * 16 * 4);
  _Float16* emb0h = (_Float16*)carve(2048L * 128 * 2);
  float*    uhat  = (float*)carve(2048L * 80 * 512 * 4);         // reused layers 0/1
  float*    vs0   = (float*)carve(2048L * 128 * 4);
  _Float16* emb1h = (_Float16*)carve(2048L * 128 * 2);
  float*    vs1   = (float*)carve(2048L * 512 * 4);
  float*    ln1o  = (float*)carve(2048L * 512 * 4);
  _Float16* wc2t  = (_Float16*)carve(64L * 576 * 2);
  _Float16* wpt   = (_Float16*)carve(16L * 1280 * 2);
  _Float16* w0h   = (_Float16*)carve(80L * 128 * 8 * 2);
  _Float16* w1h   = (_Float16*)carve(80L * 512 * 8 * 2);

  // weight prep (f16 / transposed-to-[N][K])
  cast_h<<<CDIV(81920, 256), 256, 0, stream>>>(W0, w0h, 81920);   // [80,128,8] same linear layout
  cast_h<<<CDIV(327680, 256), 256, 0, stream>>>(W1, w1h, 327680); // [80,512,8]
  transpose_c2<<<CDIV(64 * 576, 256), 256, 0, stream>>>(conv2_w, wc2t);
  transpose_proj<<<CDIV(16 * 1280, 256), 256, 0, stream>>>(proj_w, wpt);

  // frontend
  conv1_relu<<<CDIV(4L * 1024 * 40 * 64, 256), 256, 0, stream>>>(inputs, conv1_w, conv1_b, x1h);
  im2col2<<<CDIV(40960L * 576, 256), 256, 0, stream>>>(x1h, acol);
  gemm_wmma_f16<<<CDIV(2560 * 4, 4), 128, 0, stream>>>(acol, wc2t, c2, 40960, 64, 576, 4, 2560 * 4);
  bias_relu_h<<<CDIV(2621440L, 256), 256, 0, stream>>>(c2, conv2_b, x2h, 2621440L, 64);
  gemm_wmma_f16<<<CDIV(128, 4), 128, 0, stream>>>(x2h, wpt, projC, 2048, 16, 1280, 1, 128);
  bias_add_f<<<CDIV(32768L, 256), 256, 0, stream>>>(projC, proj_b, embp, 32768L, 16);
  embcaps<<<CDIV(2048, 256), 256, 0, stream>>>(embp, ec0_w, ec0_b, ec1_w, ec1_b,
                                               lni_g, lni_b, lens, emb0h);

  // routing layer 0: u_hat [2048,80,128], CO=16, CD=8, LDS-staged slice
  uhat_wmma<<<dim3(8, 128, 80), 32, 0, stream>>>(emb0h, w0h, B0, uhat, 128);
  route_scan<<<4, 512, 0, stream>>>(uhat, vs0, 16, 8, 0, 1);
  ln_rows<<<CDIV(2048, 256), 256, 0, stream>>>(vs0, lnm0_g, lnm0_b, nullptr, emb1h, 2048, 128);

  // routing layer 1: u_hat [2048,80,512], CO=32, CD=16, mask first capsule
  uhat_wmma<<<dim3(32, 128, 80), 32, 0, stream>>>(emb1h, w1h, B1, uhat, 512);
  route_scan<<<4, 512, 0, stream>>>(uhat, vs1, 32, 16, 1, 0);
  ln_rows<<<CDIV(2048, 256), 256, 0, stream>>>(vs1, lnm1_g, lnm1_b, ln1o, nullptr, 2048, 512);

  // capsule lengths + final LN(32)
  final_len_ln<<<CDIV(2048, 256), 256, 0, stream>>>(ln1o, lno_g, lno_b, out);
}
